// Encoder_84748294685032
// MI455X (gfx1250) — matrix operand order repaired and checked by reading, not compiled, not run on hardware
//
#include <hip/hip_runtime.h>
#include <hip/hip_bf16.h>

// ---------------- problem constants ----------------
#define SEQ   512
#define BATCH 64
#define INDIM 300
#define HID   512
#define GDIM  2048       // 4*HID
#define KX    320        // input K padded to multiple of 32
#define KSX   10         // x-projection k-steps (320/32)
#define KSH   16         // recurrent k-steps (512/32)
#define NTIL  128        // N tiles (2048/16)
#define MTIL  4          // M tiles (64/16)

typedef __attribute__((ext_vector_type(16))) __bf16 v16bf;
typedef __attribute__((ext_vector_type(8)))  float  v8f;

// ---------------- workspace layout (bytes) ----------------
static constexpr size_t OFF_CTR  = 0;                                    // 256 B
static constexpr size_t OFF_BSUM = 256;                                  // 2048 f32
static constexpr size_t OFF_HFR  = OFF_BSUM + (size_t)GDIM * 4;          // h double buffer
static constexpr size_t HFR_ELEMS = (size_t)2 * MTIL * KSH * 32 * 16;    // 65536 bf16
static constexpr size_t OFF_WHH  = OFF_HFR + HFR_ELEMS * 2;
static constexpr size_t WHH_ELEMS = (size_t)NTIL * KSH * 32 * 16;        // 1,048,576
static constexpr size_t OFF_WIH  = OFF_WHH + WHH_ELEMS * 2;
static constexpr size_t WIH_ELEMS = (size_t)NTIL * KSX * 32 * 16;        // 655,360
static constexpr size_t OFF_XFR  = OFF_WIH + WIH_ELEMS * 2;
static constexpr size_t XFR_ELEMS = (size_t)SEQ * MTIL * KSX * 32 * 16;  // 10,485,760

// ---------------- activation helpers ----------------
__device__ __forceinline__ float enc_sig(float x) {
    return 1.0f / (1.0f + __expf(-x));
}
__device__ __forceinline__ float enc_tanh(float x) {
    float ax = fabsf(x);
    float e  = __expf(-2.0f * ax);
    float t  = (1.0f - e) / (1.0f + e);
    return copysignf(t, x);
}

// ---------------- phase 0: init barrier, combined bias, zero h ----------------
__global__ void enc_init(unsigned* __restrict__ ctr, float* __restrict__ bsum,
                         __bf16* __restrict__ hfr,
                         const float* __restrict__ b_ih, const float* __restrict__ b_hh) {
    int i = blockIdx.x * blockDim.x + threadIdx.x;           // grid covers 65536
    if (i < (int)HFR_ELEMS) hfr[i] = (__bf16)0.0f;
    if (i < GDIM)           bsum[i] = b_ih[i] + b_hh[i];
    if (i == 0)             *ctr = 0u;
}

// ---------------- phase 0: pack W_hh into B-fragment layout ----------------
// B frag (32x16, bf16): lane = (kh<<4)|n ; elem e -> K = 32*ks + 8*kh + e + (e&8 ? 8 : 0) ; B[k][n]=W[n_g][k]
__global__ void enc_pack_whh(__bf16* __restrict__ dst, const float* __restrict__ W) {
    int i = blockIdx.x * blockDim.x + threadIdx.x;           // 1,048,576 elems
    int e = i & 15, lane = (i >> 4) & 31, ks = (i >> 9) & 15, nt = i >> 13;
    int k = 32 * ks + 8 * (lane >> 4) + e + ((e & 8) ? 8 : 0);
    int g = nt * 16 + (lane & 15);
    dst[i] = (__bf16)W[(size_t)g * HID + k];
}

__global__ void enc_pack_wih(__bf16* __restrict__ dst, const float* __restrict__ W) {
    int i = blockIdx.x * blockDim.x + threadIdx.x;           // 655,360 elems
    if (i >= (int)WIH_ELEMS) return;
    int e = i & 15, lane = (i >> 4) & 31, r = i >> 9;
    int ks = r % KSX, nt = r / KSX;
    int k = 32 * ks + 8 * (lane >> 4) + e + ((e & 8) ? 8 : 0);
    int g = nt * 16 + (lane & 15);
    float v = (k < INDIM) ? W[(size_t)g * INDIM + k] : 0.0f;
    dst[i] = (__bf16)v;
}

// ---------------- phase 0: pack inputs into A-fragment layout ----------------
// A frag (16x32, bf16): lane = (hi<<4)|m ; elem e -> K = 32*ks + 8*hi + e + (e&8 ? 8 : 0)
__global__ void enc_pack_x(__bf16* __restrict__ dst, const float* __restrict__ x) {
    int i = blockIdx.x * blockDim.x + threadIdx.x;           // 10,485,760 elems
    int e = i & 15, lane = (i >> 4) & 31, r = i >> 9;
    int ks = r % KSX; r /= KSX;
    int mt = r & 3;  int t = r >> 2;
    int hi = lane >> 4;
    int k = 32 * ks + 8 * hi + e + ((e & 8) ? 8 : 0);
    int b = mt * 16 + (lane & 15);
    float v = (k < INDIM) ? x[((size_t)t * BATCH + b) * INDIM + k] : 0.0f;
    dst[i] = (__bf16)v;
}

// ---------------- persistent LSTM recurrence ----------------
// 128 blocks x 128 threads. Block (mt,jt) owns batch rows [mt*16,+16) and hidden
// cols [jt*16,+16); computes the 4 gate tiles i/f/g/o. Split-K over 4 waves,
// ALL weight B-fragments pinned in VGPRs for the entire sequence; all A loads
// issued before the WMMA block; next-step x fragments prefetched before the
// grid barrier; LDS float4 reduction; wave 0 keeps fp32 cell state in
// registers and scatters h into next step's A-fragment layout (double buffer,
// one grid sync per step).
__global__ __launch_bounds__(128, 1)
void enc_lstm(const __bf16* __restrict__ xfr,
              const __bf16* __restrict__ whhF,
              const __bf16* __restrict__ wihF,
              __bf16* __restrict__ hfr,
              const float* __restrict__ bsum,
              unsigned* __restrict__ ctr,
              float* __restrict__ out) {
    const int blk  = blockIdx.x;
    const int mt   = blk >> 5;          // 0..3
    const int jt   = blk & 31;          // 0..31
    const int tid  = threadIdx.x;
    const int lane = tid & 31;
    const int wave = __builtin_amdgcn_readfirstlane(tid >> 5);  // uniform 0..3
    const int nl   = lane & 15;
    const int hi   = lane >> 4;

    __shared__ float red[3][4][32][8];  // 12 KB partial-C exchange

    // ---- pin all weight B-fragments in registers (loop invariant) ----
    v16bf bh[4][4];                     // recurrent: ks = wave + 4*ii, gate g
    #pragma unroll
    for (int ii = 0; ii < 4; ++ii) {
        const int ks = wave + 4 * ii;
        #pragma unroll
        for (int g = 0; g < 4; ++g) {
            const int nt = jt + 32 * g;
            bh[ii][g] = *reinterpret_cast<const v16bf*>(
                whhF + (((size_t)nt * KSH + ks) * 32 + lane) * 16);
        }
    }
    v16bf bx[3][4];                     // x-proj: s = wave + 4*jj (valid if < 10)
    #pragma unroll
    for (int jj = 0; jj < 3; ++jj) {
        const int s = wave + 4 * jj;
        if (s < KSX) {                  // wave-uniform -> scalar branch
            #pragma unroll
            for (int g = 0; g < 4; ++g) {
                const int nt = jt + 32 * g;
                bx[jj][g] = *reinterpret_cast<const v16bf*>(
                    wihF + (((size_t)nt * KSX + s) * 32 + lane) * 16);
            }
        }
    }

    // fp32 cell state lives in registers for the whole sequence (wave 0 uses it)
    float c0[8];
    #pragma unroll
    for (int r = 0; r < 8; ++r) c0[r] = 0.0f;

    const int hid = jt * 16 + nl;
    const float bi  = bsum[0 * HID + hid];
    const float bff = bsum[1 * HID + hid];
    const float bg  = bsum[2 * HID + hid];
    const float bo  = bsum[3 * HID + hid];

    // where this lane's h values land in next step's A-fragment layout
    const int ksn = hid >> 5;
    const int k32 = hid & 31;
    const int hi2 = (k32 >> 3) & 1;
    const int e2  = (k32 & 7) + ((k32 >> 4) & 1) * 8;

    // A-fragment base pointers (x base advances one timestep per iteration)
    const __bf16* xbase = xfr + (((size_t)mt * KSX + wave) * 32 + lane) * 16;
    const size_t  xstep = (size_t)MTIL * KSX * 32 * 16;   // elems per t
    const __bf16* hbase = hfr + (((size_t)mt * KSH + wave) * 32 + lane) * 16;
    const size_t  hbufs = (size_t)MTIL * KSH * 32 * 16;   // buffer stride

    for (int t = 0; t < SEQ; ++t) {
        const int buf = t & 1;
        const __bf16* hb = hbase + (size_t)buf * hbufs;

        // ---- issue ALL A-fragment loads up front (overlap with WMMA issue) ----
        v16bf afr[7];
        #pragma unroll
        for (int jj = 0; jj < 3; ++jj) {
            if (wave + 4 * jj < KSX)    // wave-uniform
                afr[jj] = *reinterpret_cast<const v16bf*>(xbase + (size_t)jj * 4 * 512);
        }
        #pragma unroll
        for (int ii = 0; ii < 4; ++ii)
            afr[3 + ii] = *reinterpret_cast<const v16bf*>(hb + (size_t)ii * 4 * 512);

        // ---- 28 WMMAs into persistent accumulators ----
        v8f acc[4];
        #pragma unroll
        for (int g = 0; g < 4; ++g) acc[g] = (v8f){0.f,0.f,0.f,0.f,0.f,0.f,0.f,0.f};
        #pragma unroll
        for (int jj = 0; jj < 3; ++jj) {
            if (wave + 4 * jj < KSX) {
                #pragma unroll
                for (int g = 0; g < 4; ++g)
                    acc[g] = __builtin_amdgcn_wmma_f32_16x16x32_bf16(
                        false, afr[jj], false, bx[jj][g], (short)0, acc[g], false, false);
            }
        }
        #pragma unroll
        for (int ii = 0; ii < 4; ++ii) {
            #pragma unroll
            for (int g = 0; g < 4; ++g)
                acc[g] = __builtin_amdgcn_wmma_f32_16x16x32_bf16(
                    false, afr[3 + ii], false, bh[ii][g], (short)0, acc[g], false, false);
        }
        xbase += xstep;

        // ---- warm next step's x fragments while we reduce + sync ----
        if (t + 1 < SEQ) {              // uniform
            #pragma unroll
            for (int jj = 0; jj < 3; ++jj) {
                if (wave + 4 * jj < KSX)
                    __builtin_prefetch(xbase + (size_t)jj * 4 * 512, 0, 3);
            }
        }

        // ---- cross-wave reduction through LDS (float4 = ds_*_b128) ----
        if (wave > 0) {
            #pragma unroll
            for (int g = 0; g < 4; ++g) {
                float4 lo = make_float4(acc[g][0], acc[g][1], acc[g][2], acc[g][3]);
                float4 hi4 = make_float4(acc[g][4], acc[g][5], acc[g][6], acc[g][7]);
                *reinterpret_cast<float4*>(&red[wave - 1][g][lane][0]) = lo;
                *reinterpret_cast<float4*>(&red[wave - 1][g][lane][4]) = hi4;
            }
        }
        __syncthreads();

        if (wave == 0) {
            #pragma unroll
            for (int w = 0; w < 3; ++w) {
                #pragma unroll
                for (int g = 0; g < 4; ++g) {
                    float4 lo = *reinterpret_cast<const float4*>(&red[w][g][lane][0]);
                    float4 hi4 = *reinterpret_cast<const float4*>(&red[w][g][lane][4]);
                    acc[g][0] += lo.x;  acc[g][1] += lo.y;
                    acc[g][2] += lo.z;  acc[g][3] += lo.w;
                    acc[g][4] += hi4.x; acc[g][5] += hi4.y;
                    acc[g][6] += hi4.z; acc[g][7] += hi4.w;
                }
            }
            // identical C layout across the 4 accumulators => each lane holds
            // i/f/g/o for the same (b, hid) cells -> fused cell update in-register
            __bf16* hN = hfr + (size_t)(buf ^ 1) * hbufs;
            #pragma unroll
            for (int r = 0; r < 8; ++r) {
                float iv = enc_sig(acc[0][r] + bi);
                float fv = enc_sig(acc[1][r] + bff);
                float gv = enc_tanh(acc[2][r] + bg);
                float ov = enc_sig(acc[3][r] + bo);
                float cc = fv * c0[r] + iv * gv;
                c0[r] = cc;
                float hh = ov * enc_tanh(cc);
                // scatter h straight into next step's A-fragment layout
                int laneD = (r + 8 * hi) + 16 * hi2;
                hN[(((size_t)mt * KSH + ksn) * 32 + laneD) * 16 + e2] = (__bf16)hh;
                if (t == SEQ - 1) {
                    int b = mt * 16 + r + 8 * hi;
                    out[(size_t)b * HID + hid] = hh;   // final hidden state, f32
                }
            }
        }
        __syncthreads();

        // one grid-wide barrier per step: monotonic counter, no reset needed
        if (tid == 0) {
            __threadfence();
            atomicAdd(ctr, 1u);
            const unsigned target = 128u * (unsigned)(t + 1);
            while (__hip_atomic_load(ctr, __ATOMIC_RELAXED, __HIP_MEMORY_SCOPE_AGENT) < target)
                __builtin_amdgcn_s_sleep(1);
            __threadfence();
        }
        __syncthreads();
    }
}

// ---------------- host launcher ----------------
extern "C" void kernel_launch(void* const* d_in, const int* in_sizes, int n_in,
                              void* d_out, int out_size, void* d_ws, size_t ws_size,
                              hipStream_t stream) {
    const float* x    = (const float*)d_in[0];   // [512,64,300]
    const float* W_ih = (const float*)d_in[1];   // [2048,300]
    const float* W_hh = (const float*)d_in[2];   // [2048,512]
    const float* b_ih = (const float*)d_in[3];   // [2048]
    const float* b_hh = (const float*)d_in[4];   // [2048]

    char* ws = (char*)d_ws;                      // needs ~24.5 MB
    unsigned* ctr  = (unsigned*)(ws + OFF_CTR);
    float*    bsum = (float*)(ws + OFF_BSUM);
    __bf16*   hfr  = (__bf16*)(ws + OFF_HFR);
    __bf16*   whhF = (__bf16*)(ws + OFF_WHH);
    __bf16*   wihF = (__bf16*)(ws + OFF_WIH);
    __bf16*   xfr  = (__bf16*)(ws + OFF_XFR);

    enc_init<<<256, 256, 0, stream>>>(ctr, bsum, hfr, b_ih, b_hh);
    enc_pack_whh<<<(int)(WHH_ELEMS / 256), 256, 0, stream>>>(whhF, W_hh);
    enc_pack_wih<<<(int)((WIH_ELEMS + 255) / 256), 256, 0, stream>>>(wihF, W_ih);
    enc_pack_x<<<(int)(XFR_ELEMS / 256), 256, 0, stream>>>(xfr, x);
    enc_lstm<<<128, 128, 0, stream>>>(xfr, whhF, wihF, hfr, bsum, ctr, (float*)d_out);
}
